// TransformerBlock_6133213299369
// MI455X (gfx1250) — matrix operand order repaired and checked by reading, not compiled, not run on hardware
//
#include <hip/hip_runtime.h>

// ---------------------------------------------------------------------------
// Point Transformer block for MI455X (gfx1250, wave32, WMMA f16->f32).
// ---------------------------------------------------------------------------

#define BATCH 2
#define NPTS  8192
#define DP    64
#define DM    128
#define KNN   16
#define MTOT  (BATCH * NPTS)

#define SEGS   4
#define SEGLEN (NPTS / SEGS)
#define CHUNK  1024

typedef __attribute__((ext_vector_type(16))) _Float16 v16h;
typedef __attribute__((ext_vector_type(8)))  float    v8f;

// ------------------------- WMMA fragment helpers ---------------------------
// A: 16x32 f16, row-major source with row stride ld (elements).
// lane&15 = M row; lane>>4 selects K half: K = {kh*8 + e} U {16 + kh*8 + e}.
__device__ __forceinline__ v16h a_frag_f32(const float* __restrict__ base, int ld, int lane) {
    const float* r = base + (size_t)(lane & 15) * ld + ((lane >> 4) & 1) * 8;
    v16h a;
#pragma unroll
    for (int e = 0; e < 8; ++e) a[e] = (_Float16)r[e];
#pragma unroll
    for (int e = 0; e < 8; ++e) a[8 + e] = (_Float16)r[16 + e];
    return a;
}

__device__ __forceinline__ v16h a_frag_f16(const _Float16* __restrict__ base, int ld, int lane) {
    const _Float16* r = base + (size_t)(lane & 15) * ld + ((lane >> 4) & 1) * 8;
    v16h a;
#pragma unroll
    for (int e = 0; e < 8; ++e) a[e] = r[e];
#pragma unroll
    for (int e = 0; e < 8; ++e) a[8 + e] = r[16 + e];
    return a;
}

// B: 32x16 f16, K-major source (row stride ld = N). lane&15 = column N;
// K = {kh*8 + e} U {16 + kh*8 + e}.
__device__ __forceinline__ v16h b_frag_f16(const _Float16* __restrict__ base, int ld, int lane) {
    const _Float16* c = base + (lane & 15) + (size_t)((lane >> 4) & 1) * 8 * ld;
    v16h b;
#pragma unroll
    for (int e = 0; e < 8; ++e) b[e] = c[(size_t)e * ld];
#pragma unroll
    for (int e = 0; e < 8; ++e) b[8 + e] = c[(size_t)(16 + e) * ld];
    return b;
}

__device__ __forceinline__ v8f wmma16(v16h a, v16h b, v8f c) {
    return __builtin_amdgcn_wmma_f32_16x16x32_f16(false, a, false, b, (short)0, c, false, false);
}

// ------------------------------ KNN ----------------------------------------
__global__ __launch_bounds__(256) void knn_partial_kernel(
        const float* __restrict__ xyz, float* __restrict__ pdist, int* __restrict__ pidx) {
    __shared__ float sx[CHUNK * 3];
    const int qi  = blockIdx.x * 256 + threadIdx.x;
    const int b   = qi / NPTS;
    const int seg = blockIdx.y;
    const float* xb = xyz + (size_t)b * NPTS * 3;
    const float x0 = xyz[(size_t)qi * 3 + 0];
    const float x1 = xyz[(size_t)qi * 3 + 1];
    const float x2 = xyz[(size_t)qi * 3 + 2];

    float dist[KNN]; int idx[KNN];
#pragma unroll
    for (int s = 0; s < KNN; ++s) { dist[s] = 3.4e38f; idx[s] = -1; }

    const int segEnd = seg * SEGLEN + SEGLEN;
    for (int c0 = seg * SEGLEN; c0 < segEnd; c0 += CHUNK) {
        __syncthreads();
        for (int t = threadIdx.x; t < CHUNK * 3; t += 256) sx[t] = xb[(size_t)c0 * 3 + t];
        __syncthreads();
        // prefetch next chunk into cache while we grind this one (global_prefetch_b8)
        if (c0 + CHUNK < segEnd)
            __builtin_prefetch(xb + (size_t)(c0 + CHUNK) * 3 + threadIdx.x * 12, 0, 3);
        for (int j = 0; j < CHUNK; ++j) {
            float d0 = x0 - sx[j * 3 + 0];
            float d1 = x1 - sx[j * 3 + 1];
            float d2 = x2 - sx[j * 3 + 2];
            float d  = d0 * d0 + d1 * d1 + d2 * d2;
            if (d < dist[KNN - 1]) {
                float dd = d; int ii = c0 + j;
#pragma unroll
                for (int s = 0; s < KNN; ++s) {
                    if (dd < dist[s]) {
                        float td = dist[s]; int ti = idx[s];
                        dist[s] = dd; idx[s] = ii; dd = td; ii = ti;
                    }
                }
            }
        }
    }
#pragma unroll
    for (int s = 0; s < KNN; ++s) {
        pdist[((size_t)qi * SEGS + seg) * KNN + s] = dist[s];
        pidx [((size_t)qi * SEGS + seg) * KNN + s] = idx[s];
    }
}

__global__ __launch_bounds__(256) void knn_merge_kernel(
        const float* __restrict__ pdist, const int* __restrict__ pidx,
        int* __restrict__ knn_idx) {
    const int qi = blockIdx.x * 256 + threadIdx.x;
    float dist[KNN]; int idx[KNN];
#pragma unroll
    for (int s = 0; s < KNN; ++s) { dist[s] = 3.4e38f; idx[s] = -1; }
    for (int c = 0; c < SEGS * KNN; ++c) {
        float d = pdist[(size_t)qi * SEGS * KNN + c];
        int  ii = pidx [(size_t)qi * SEGS * KNN + c];
        if (d < dist[KNN - 1]) {
            float dd = d;
#pragma unroll
            for (int s = 0; s < KNN; ++s) {
                if (dd < dist[s]) {
                    float td = dist[s]; int ti = idx[s];
                    dist[s] = dd; idx[s] = ii; dd = td; ii = ti;
                }
            }
        }
    }
#pragma unroll
    for (int s = 0; s < KNN; ++s) knn_idx[(size_t)qi * KNN + s] = idx[s];
}

// -------------------- weight f32 -> f16 conversion -------------------------
// g16 layout: [fc1 8192][wq 16384][wk][wv][gw1][gw2][dw2]
#define G16_FC1 0
#define G16_WQ  (8192)
#define G16_WK  (8192 + 16384)
#define G16_WV  (8192 + 2 * 16384)
#define G16_GW1 (8192 + 3 * 16384)
#define G16_GW2 (8192 + 4 * 16384)
#define G16_DW2 (8192 + 5 * 16384)
#define G16_TOT (8192 + 6 * 16384)

__global__ __launch_bounds__(256) void cvt_kernel(
        const float* __restrict__ fc1, const float* __restrict__ wq,
        const float* __restrict__ wk,  const float* __restrict__ wv,
        const float* __restrict__ gw1, const float* __restrict__ gw2,
        const float* __restrict__ dw2, _Float16* __restrict__ g16) {
    int t = blockIdx.x * 256 + threadIdx.x;
    if (t >= G16_TOT) return;
    if (t < 8192) { g16[t] = (_Float16)fc1[t]; return; }
    int u = t - 8192;
    int seg = u / 16384, r = u % 16384;
    const float* s = (seg == 0) ? wq : (seg == 1) ? wk : (seg == 2) ? wv
                   : (seg == 3) ? gw1 : (seg == 4) ? gw2 : dw2;
    g16[t] = (_Float16)s[r];
}

// ----------------- fused fc1 + q/k/v projection kernel ---------------------
// Block owns 128 rows. Phase 1: x = feat@fc1 + b  -> LDS f16 tile (128x128).
// Phase 2: q/k/v = x@W for three weights, A-fragments straight from LDS.
__global__ __launch_bounds__(256) void proj_kernel(
        const float* __restrict__ feat, const _Float16* __restrict__ g16,
        const float* __restrict__ fc1_b,
        float* __restrict__ q, float* __restrict__ xk, float* __restrict__ xv) {
    __shared__ _Float16 s_x[128 * DM];   // 32 KB

    const int wave = threadIdx.x >> 5, lane = threadIdx.x & 31;
    const int mbase = blockIdx.x * 128;
    const int col = wave * 16 + (lane & 15);
    const _Float16* fc1w = g16 + G16_FC1;
    const _Float16* wq16 = g16 + G16_WQ;
    const _Float16* wk16 = g16 + G16_WK;
    const _Float16* wv16 = g16 + G16_WV;

    // Phase 1: x tile into LDS (f16)
    const float bv = fc1_b[col];
    for (int ms = 0; ms < 8; ++ms) {
        const int row0 = mbase + ms * 16;
        v8f acc = {};
#pragma unroll
        for (int ks = 0; ks < DP / 32; ++ks) {
            v16h a = a_frag_f32(feat + (size_t)row0 * DP + ks * 32, DP, lane);
            v16h b = b_frag_f16(fc1w + (size_t)ks * 32 * DM + wave * 16, DM, lane);
            acc = wmma16(a, b, acc);
        }
#pragma unroll
        for (int r = 0; r < 8; ++r) {
            int lrow = ms * 16 + r + 8 * (lane >> 4);
            s_x[lrow * DM + col] = (_Float16)(acc[r] + bv);
        }
    }
    __syncthreads();

    // Phase 2: q/k/v from LDS x tile
    for (int ms = 0; ms < 8; ++ms) {
        const int row0 = mbase + ms * 16;
        v8f aq = {}, ak = {}, av = {};
#pragma unroll
        for (int ks = 0; ks < DM / 32; ++ks) {
            v16h a = a_frag_f16(s_x + (size_t)ms * 16 * DM + ks * 32, DM, lane);
            aq = wmma16(a, b_frag_f16(wq16 + (size_t)ks * 32 * DM + wave * 16, DM, lane), aq);
            ak = wmma16(a, b_frag_f16(wk16 + (size_t)ks * 32 * DM + wave * 16, DM, lane), ak);
            av = wmma16(a, b_frag_f16(wv16 + (size_t)ks * 32 * DM + wave * 16, DM, lane), av);
        }
#pragma unroll
        for (int r = 0; r < 8; ++r) {
            size_t o = (size_t)(row0 + r + 8 * (lane >> 4)) * DM + col;
            q[o] = aq[r]; xk[o] = ak[r]; xv[o] = av[r];
        }
    }
}

// --------------------- fused per-query attention kernel --------------------
// One query per M-tile: 16 neighbor rows x 128 channels.
// delta-l2, gamma-l1, gamma-l2 are WMMA; softmax over K per channel; fc2 GEMV.
#define QPB 16   // queries per block; grid = MTOT/QPB = 1024

__global__ __launch_bounds__(256) void attn_kernel(
        const float* __restrict__ xyz, const float* __restrict__ feat,
        const float* __restrict__ qm,  const float* __restrict__ xk,
        const float* __restrict__ xv,  const int* __restrict__ knn,
        const _Float16* __restrict__ g16,
        const float* __restrict__ dw1, const float* __restrict__ db1,
        const float* __restrict__ db2, const float* __restrict__ gb1,
        const float* __restrict__ gb2,
        const float* __restrict__ fc2w, const float* __restrict__ fc2b,
        float* __restrict__ out_res, float* __restrict__ out_attn) {
    __shared__ float    s_dw1[3 * DM];
    __shared__ float    s_db1[DM], s_db2[DM], s_gb1[DM], s_gb2[DM];
    __shared__ int      s_idx[KNN];
    __shared__ float    s_rel[KNN * 3];
    __shared__ _Float16 s_pe1[KNN * DM];
    __shared__ float    s_pos[KNN * DM];
    __shared__ _Float16 s_h[KNN * DM];
    __shared__ _Float16 s_g1[KNN * DM];
    __shared__ float    s_logit[KNN * DM];
    __shared__ float    s_res[DM];

    const _Float16* gw1_16 = g16 + G16_GW1;
    const _Float16* gw2_16 = g16 + G16_GW2;
    const _Float16* dw2_16 = g16 + G16_DW2;

    for (int t = threadIdx.x; t < 3 * DM; t += 256) s_dw1[t] = dw1[t];
    for (int t = threadIdx.x; t < DM; t += 256) {
        s_db1[t] = db1[t]; s_db2[t] = db2[t]; s_gb1[t] = gb1[t]; s_gb2[t] = gb2[t];
    }

    const int wave = threadIdx.x >> 5, lane = threadIdx.x & 31;
    const int colw = wave * 16 + (lane & 15);
    const int rowh = 8 * (lane >> 4);

    for (int it = 0; it < QPB; ++it) {
        const int qi = blockIdx.x * QPB + it;
        const int b = qi / NPTS;
        __syncthreads();                       // LDS reuse fence (also covers bias staging)
        if (threadIdx.x < KNN) s_idx[threadIdx.x] = knn[(size_t)qi * KNN + threadIdx.x];
        __syncthreads();
        if (threadIdx.x < KNN * 3) {
            int k = threadIdx.x / 3, c = threadIdx.x % 3;
            s_rel[threadIdx.x] = xyz[(size_t)qi * 3 + c]
                               - xyz[((size_t)b * NPTS + s_idx[k]) * 3 + c];
        }
        __syncthreads();
        // delta layer 1: relu(rel @ dw1 + db1)  (tiny: VALU)
        for (int t = threadIdx.x; t < KNN * DM; t += 256) {
            int k = t >> 7, f = t & 127;
            float v = s_rel[k * 3 + 0] * s_dw1[f]
                    + s_rel[k * 3 + 1] * s_dw1[DM + f]
                    + s_rel[k * 3 + 2] * s_dw1[2 * DM + f] + s_db1[f];
            s_pe1[t] = (_Float16)fmaxf(v, 0.f);
        }
        __syncthreads();
        // delta layer 2 (WMMA) -> pos_enc; build h = q - k_nbr + pos_enc
        {
            v8f acc = {};
#pragma unroll
            for (int ks = 0; ks < 4; ++ks)
                acc = wmma16(a_frag_f16(s_pe1 + ks * 32, DM, lane),
                             b_frag_f16(dw2_16 + (size_t)ks * 32 * DM + wave * 16, DM, lane), acc);
            const float bias = s_db2[colw];
            const float qv = qm[(size_t)qi * DM + colw];
#pragma unroll
            for (int r = 0; r < 8; ++r) {
                int row = r + rowh;
                float pos = fmaxf(acc[r] + bias, 0.f);
                s_pos[row * DM + colw] = pos;
                float kv = xk[((size_t)b * NPTS + s_idx[row]) * DM + colw];
                s_h[row * DM + colw] = (_Float16)(qv - kv + pos);
            }
        }
        __syncthreads();
        // gamma layer 1 (WMMA) + relu
        {
            v8f acc = {};
#pragma unroll
            for (int ks = 0; ks < 4; ++ks)
                acc = wmma16(a_frag_f16(s_h + ks * 32, DM, lane),
                             b_frag_f16(gw1_16 + (size_t)ks * 32 * DM + wave * 16, DM, lane), acc);
            const float bias = s_gb1[colw];
#pragma unroll
            for (int r = 0; r < 8; ++r)
                s_g1[(r + rowh) * DM + colw] = (_Float16)fmaxf(acc[r] + bias, 0.f);
        }
        __syncthreads();
        // gamma layer 2 (WMMA) -> logits
        {
            v8f acc = {};
#pragma unroll
            for (int ks = 0; ks < 4; ++ks)
                acc = wmma16(a_frag_f16(s_g1 + ks * 32, DM, lane),
                             b_frag_f16(gw2_16 + (size_t)ks * 32 * DM + wave * 16, DM, lane), acc);
            const float bias = s_gb2[colw];
#pragma unroll
            for (int r = 0; r < 8; ++r)
                s_logit[(r + rowh) * DM + colw] = acc[r] + bias;
        }
        __syncthreads();
        // softmax over K per channel + weighted sum of (v_nbr + pos_enc)
        if (threadIdx.x < DM) {
            const int f = threadIdx.x;
            const float sc = 0.08838834764831845f;   // 1/sqrt(128)
            float m = -3.4e38f;
#pragma unroll
            for (int k = 0; k < KNN; ++k) m = fmaxf(m, s_logit[k * DM + f] * sc);
            float ex[KNN]; float sum = 0.f;
#pragma unroll
            for (int k = 0; k < KNN; ++k) {
                ex[k] = __expf(s_logit[k * DM + f] * sc - m);
                sum += ex[k];
            }
            const float inv = 1.f / sum;
            float r = 0.f;
#pragma unroll
            for (int k = 0; k < KNN; ++k) {
                float a = ex[k] * inv;
                out_attn[((size_t)qi * KNN + k) * DM + f] = a;
                float vv = xv[((size_t)b * NPTS + s_idx[k]) * DM + f] + s_pos[k * DM + f];
                r += a * vv;
            }
            s_res[f] = r;
        }
        __syncthreads();
        // fc2 + bias + residual
        if (threadIdx.x < DP) {
            const int d = threadIdx.x;
            float o = fc2b[d] + feat[(size_t)qi * DP + d];
#pragma unroll 8
            for (int f = 0; f < DM; ++f) o += s_res[f] * fc2w[f * DP + d];
            out_res[(size_t)qi * DP + d] = o;
        }
    }
}

// ------------------------------ launcher -----------------------------------
extern "C" void kernel_launch(void* const* d_in, const int* in_sizes, int n_in,
                              void* d_out, int out_size, void* d_ws, size_t ws_size,
                              hipStream_t stream) {
    const float* xyz   = (const float*)d_in[0];
    // d_in[1] = normals (unused by reference)
    const float* feat  = (const float*)d_in[2];
    const float* fc1_w = (const float*)d_in[3];
    const float* fc1_b = (const float*)d_in[4];
    const float* fc2_w = (const float*)d_in[5];
    const float* fc2_b = (const float*)d_in[6];
    const float* gw1   = (const float*)d_in[7];
    const float* gb1   = (const float*)d_in[8];
    const float* gw2   = (const float*)d_in[9];
    const float* gb2   = (const float*)d_in[10];
    const float* dw1   = (const float*)d_in[11];
    const float* db1   = (const float*)d_in[12];
    const float* dw2   = (const float*)d_in[13];
    const float* db2   = (const float*)d_in[14];
    const float* wq    = (const float*)d_in[15];
    const float* wk    = (const float*)d_in[16];
    const float* wv    = (const float*)d_in[17];

    char* ws = (char*)d_ws;
    size_t off = 0;
    auto alloc = [&](size_t bytes) -> void* {
        void* p = ws + off;
        off += (bytes + 255) & ~(size_t)255;
        return p;
    };

    int*      knn_idx = (int*)     alloc((size_t)MTOT * KNN * sizeof(int));
    float*    q       = (float*)   alloc((size_t)MTOT * DM * sizeof(float));
    float*    xk      = (float*)   alloc((size_t)MTOT * DM * sizeof(float));
    float*    xv      = (float*)   alloc((size_t)MTOT * DM * sizeof(float));
    _Float16* g16     = (_Float16*)alloc((size_t)G16_TOT * sizeof(_Float16));
    // KNN partial results only live before proj writes q/xk -> alias that space.
    float* pd = q;           // MTOT*SEGS*KNN floats = 4 MB (q region is 8 MB)
    int*   pi = (int*)xk;    // MTOT*SEGS*KNN ints  = 4 MB (xk region is 8 MB)

    float* out_res  = (float*)d_out;
    float* out_attn = (float*)d_out + (size_t)MTOT * DP;

    knn_partial_kernel<<<dim3(MTOT / 256, SEGS), 256, 0, stream>>>(xyz, pd, pi);
    knn_merge_kernel<<<MTOT / 256, 256, 0, stream>>>(pd, pi, knn_idx);
    cvt_kernel<<<(G16_TOT + 255) / 256, 256, 0, stream>>>(fc1_w, wq, wk, wv, gw1, gw2, dw2, g16);
    proj_kernel<<<MTOT / 128, 256, 0, stream>>>(feat, g16, fc1_b, q, xk, xv);
    attn_kernel<<<MTOT / QPB, 256, 0, stream>>>(xyz, feat, q, xk, xv, knn_idx, g16,
                                                dw1, db1, db2, gb1, gb2, fc2_w, fc2_b,
                                                out_res, out_attn);
}
